// MoEClusteredAttention_40089224741574
// MI455X (gfx1250) — matrix operand order repaired and checked by reading, not compiled, not run on hardware
//
#include <hip/hip_runtime.h>

// ---------------------------------------------------------------------------
// MoE clustered attention for MI455X (gfx1250), wave32 + WMMA bf16.
// B=2, LQ=LK=2048, D=1024, H=16, DH=64, M(route)=8
// ---------------------------------------------------------------------------

#define B_  2
#define LQ_ 2048
#define LK_ 2048
#define D_  1024
#define H_  16
#define DH_ 64
#define MR_ 8

typedef __attribute__((ext_vector_type(16))) __bf16 bf16x16;
typedef __attribute__((ext_vector_type(8)))  __bf16 bf16x8;
typedef __attribute__((ext_vector_type(8)))  float  f32x8;
typedef __attribute__((ext_vector_type(4)))  int    i32x4;

#if defined(__has_builtin)
#if __has_builtin(__builtin_amdgcn_global_load_async_to_lds_b128)
#define HAVE_ASYNC_LDS 1
#endif
#endif

static __device__ inline __bf16 f2bf(float f) {
    unsigned u = __builtin_bit_cast(unsigned, f);
    unsigned r = u + 0x7FFFu + ((u >> 16) & 1u);   // round-to-nearest-even
    unsigned short h = (unsigned short)(r >> 16);
    return __builtin_bit_cast(__bf16, h);
}

static __device__ inline f32x8 wmma_bf16(bf16x16 a, bf16x16 b, f32x8 c) {
    return __builtin_amdgcn_wmma_f32_16x16x32_bf16(
        /*neg_a=*/false, a, /*neg_b=*/false, b,
        /*c_mod=*/(short)0, c, /*reuse_a=*/false, /*reuse_b=*/false);
}

// A-fragment (16x32 bf16): lanes0-15 K{0..7,16..23}, lanes16-31 K{8..15,24..31}
static __device__ inline bf16x16 load_a_frag(const __bf16* row, int half) {
    bf16x8 lo = *(const bf16x8*)(row + half * 8);
    bf16x8 hi = *(const bf16x8*)(row + 16 + half * 8);
    bf16x16 r;
#pragma unroll
    for (int i = 0; i < 8; ++i) { r[i] = lo[i]; r[i + 8] = hi[i]; }
    return r;
}

// B-fragment (32x16 bf16) from transposed tile T[n][kk]: lane holds column n,
// lanes0-15 kk {0..7,16..23}, lanes16-31 kk {8..15,24..31}
static __device__ inline bf16x16 load_b_frag(const __bf16* trow, int half) {
    bf16x8 lo = *(const bf16x8*)(trow + half * 8);
    bf16x8 hi = *(const bf16x8*)(trow + 16 + half * 8);
    bf16x16 r;
#pragma unroll
    for (int i = 0; i < 8; ++i) { r[i] = lo[i]; r[i + 8] = hi[i]; }
    return r;
}

static __device__ inline float rmax16(float v) {
#pragma unroll
    for (int m = 1; m < 16; m <<= 1) v = fmaxf(v, __shfl_xor(v, m, 16));
    return v;
}
static __device__ inline float rsum16(float v) {
#pragma unroll
    for (int m = 1; m < 16; m <<= 1) v += __shfl_xor(v, m, 16);
    return v;
}

// 16-byte global -> LDS copy, async (ASYNCcnt / gfx1250 path) when available.
// Builtin prototype (from clang diagnostic): arg0 = AS1 pointer to <4 x i32>.
static __device__ inline void copy16_g2l(__bf16* dst, const __bf16* src) {
#ifdef HAVE_ASYNC_LDS
    __attribute__((address_space(1))) i32x4* g =
        (__attribute__((address_space(1))) i32x4*)(uintptr_t)src;
    __attribute__((address_space(3))) i32x4* l =
        (__attribute__((address_space(3))) i32x4*)(unsigned)(uintptr_t)dst;
    __builtin_amdgcn_global_load_async_to_lds_b128(g, l, /*offset=*/0, /*cpol=*/0);
#else
    *(bf16x8*)dst = *(const bf16x8*)src;
#endif
}

static __device__ inline void wait_async_done() {
#ifdef HAVE_ASYNC_LDS
#if __has_builtin(__builtin_amdgcn_s_wait_asynccnt)
    __builtin_amdgcn_s_wait_asynccnt(0);
#else
    asm volatile("s_wait_asynccnt 0" ::: "memory");
#endif
#endif
}

// ---------------------------------------------------------------------------
// 1) Router: assign[t] = argmax_m ( X[t,:] @ Wr[:,m] ),  first-max tiebreak
// ---------------------------------------------------------------------------
__global__ void router_kernel(const float* __restrict__ X,
                              const float* __restrict__ Wr,
                              int* __restrict__ assign, int tokens) {
    int t = blockIdx.x * blockDim.x + threadIdx.x;
    if (t >= tokens) return;
    const float* x = X + (size_t)t * D_;
    float acc[MR_];
#pragma unroll
    for (int m = 0; m < MR_; ++m) acc[m] = 0.f;
    for (int dd = 0; dd < D_; ++dd) {
        float xv = x[dd];
        const float4* w = (const float4*)(Wr + (size_t)dd * MR_);
        float4 w0 = w[0], w1 = w[1];
        acc[0] += xv * w0.x; acc[1] += xv * w0.y;
        acc[2] += xv * w0.z; acc[3] += xv * w0.w;
        acc[4] += xv * w1.x; acc[5] += xv * w1.y;
        acc[6] += xv * w1.z; acc[7] += xv * w1.w;
    }
    float best = acc[0]; int bi = 0;
#pragma unroll
    for (int m = 1; m < MR_; ++m)
        if (acc[m] > best) { best = acc[m]; bi = m; }
    assign[t] = bi;
}

// ---------------------------------------------------------------------------
// 2) WMMA GEMM:  Y[M,N] = A[M,1024](f32) @ W[1024,1024](f32), bf16/f32-acc
//    MODE 0: Y -> bf16 [B,H,L,DH]      (q, k)
//    MODE 1: Y -> bf16 [B,H,DH,L]      (v transposed)
//    MODE 2: Y -> f32  [M,N] row-major (final output)
//    Block: 256 thr (8 waves), tile 128(M) x 64(N), K-step 32.
// ---------------------------------------------------------------------------
#define GA_STR 40   // LDS stride (bf16 elems) for A tile rows of 32
#define GW_STR 40   // LDS stride for transposed W tile rows of 32

template <int MODE>
__global__ void __launch_bounds__(256)
wmma_gemm_kernel(const float* __restrict__ A, const float* __restrict__ W,
                 void* __restrict__ out) {
    constexpr int K = D_, N = D_, L = LQ_;

    __shared__ __bf16 At[128 * GA_STR];
    __shared__ __bf16 Wt[64 * GW_STR];

    const int tid  = threadIdx.x;
    const int wave = tid >> 5;
    const int lane = tid & 31;
    const int half = lane >> 4;
    const int r    = lane & 15;

    const int m0 = blockIdx.y * 128;
    const int n0 = blockIdx.x * 64;

    const int ar = tid >> 1, ks = (tid & 1) * 16;       // A-tile loader coords
    const int wkk = tid >> 3, wnb = (tid & 7) * 8;      // W-tile loader coords

    f32x8 acc[4];
#pragma unroll
    for (int d = 0; d < 4; ++d) acc[d] = (f32x8){0,0,0,0,0,0,0,0};

    for (int k0 = 0; k0 < K; k0 += 32) {
        __syncthreads();
        // A tile: thread t loads 16 consecutive K-floats of one row
        {
            const float4* ap = (const float4*)(A + (size_t)(m0 + ar) * K + k0 + ks);
            float4 a0 = ap[0], a1 = ap[1], a2 = ap[2], a3 = ap[3];
            __bf16* dst = &At[ar * GA_STR + ks];
            dst[0]=f2bf(a0.x); dst[1]=f2bf(a0.y); dst[2]=f2bf(a0.z); dst[3]=f2bf(a0.w);
            dst[4]=f2bf(a1.x); dst[5]=f2bf(a1.y); dst[6]=f2bf(a1.z); dst[7]=f2bf(a1.w);
            dst[8]=f2bf(a2.x); dst[9]=f2bf(a2.y); dst[10]=f2bf(a2.z); dst[11]=f2bf(a2.w);
            dst[12]=f2bf(a3.x); dst[13]=f2bf(a3.y); dst[14]=f2bf(a3.z); dst[15]=f2bf(a3.w);
        }
        // W tile, stored transposed: Wt[n][kk] = W[k0+kk][n0+n]
        {
            const float4* wp = (const float4*)(W + (size_t)(k0 + wkk) * N + n0 + wnb);
            float4 w0 = wp[0], w1 = wp[1];
            Wt[(wnb+0)*GW_STR + wkk] = f2bf(w0.x);
            Wt[(wnb+1)*GW_STR + wkk] = f2bf(w0.y);
            Wt[(wnb+2)*GW_STR + wkk] = f2bf(w0.z);
            Wt[(wnb+3)*GW_STR + wkk] = f2bf(w0.w);
            Wt[(wnb+4)*GW_STR + wkk] = f2bf(w1.x);
            Wt[(wnb+5)*GW_STR + wkk] = f2bf(w1.y);
            Wt[(wnb+6)*GW_STR + wkk] = f2bf(w1.z);
            Wt[(wnb+7)*GW_STR + wkk] = f2bf(w1.w);
        }
        // prefetch next K-tile into cache while this one computes
        if (k0 + 32 < K) {
            __builtin_prefetch(A + (size_t)(m0 + ar) * K + k0 + 32 + ks, 0, 1);
            __builtin_prefetch(W + (size_t)(k0 + 32 + wkk) * N + n0 + wnb, 0, 1);
        }
        __syncthreads();

        bf16x16 afrag = load_a_frag(&At[(wave * 16 + r) * GA_STR], half);
#pragma unroll
        for (int d = 0; d < 4; ++d) {
            bf16x16 bfrag = load_b_frag(&Wt[(d * 16 + r) * GW_STR], half);
            acc[d] = wmma_bf16(afrag, bfrag, acc[d]);
        }
    }

    // Epilogue: C element j <-> row (j + 8*half), col (d*16 + r)
#pragma unroll
    for (int d = 0; d < 4; ++d) {
#pragma unroll
        for (int j = 0; j < 8; ++j) {
            int gr = m0 + wave * 16 + j + 8 * half;
            int gc = n0 + d * 16 + r;
            float v = acc[d][j];
            if constexpr (MODE == 2) {
                ((float*)out)[(size_t)gr * N + gc] = v;
            } else {
                int b = gr / L, l = gr % L;        // L power-of-2 -> shifts
                int h = gc / DH_, dh = gc % DH_;
                __bf16* o = (__bf16*)out;
                if constexpr (MODE == 0)
                    o[(((size_t)b * H_ + h) * L + l) * DH_ + dh] = f2bf(v);
                else // transposed value store
                    o[(((size_t)b * H_ + h) * DH_ + dh) * L + l] = f2bf(v);
            }
        }
    }
}

// ---------------------------------------------------------------------------
// 3) Clustered flash attention. 8 waves/block; wave owns a 16-query tile of
//    one (b,h). KV processed in 32-key steps. ctx written f32 [B,LQ,D]
//    (head-major cols) so the Wo GEMM consumes it directly.
// ---------------------------------------------------------------------------
#define KT_STR 72   // K tile rows of DH=64 (+8 pad)
#define VT_STR 40   // V^T tile rows of 32 keys (+8 pad)
#define PB_STR 40   // per-wave P buffer rows of 32 keys (+8 pad)

__global__ void __launch_bounds__(256)
attention_kernel(const __bf16* __restrict__ qw,   // [B,H,LQ,DH]
                 const __bf16* __restrict__ kw,   // [B,H,LK,DH]
                 const __bf16* __restrict__ vtw,  // [B,H,DH,LK]
                 const int* __restrict__ assign_q,
                 const int* __restrict__ assign_k,
                 float* __restrict__ ctx) {       // [B,LQ,D]
    __shared__ __bf16 Kt[32 * KT_STR];
    __shared__ __bf16 Vt[64 * VT_STR];
    __shared__ __bf16 Pb[8 * 16 * PB_STR];

    const int tid  = threadIdx.x;
    const int wave = tid >> 5;
    const int lane = tid & 31;
    const int half = lane >> 4;
    const int r    = lane & 15;

    const int b = blockIdx.z, h = blockIdx.y;
    const int qtile = blockIdx.x * 128 + wave * 16;
    const float SCALE = 0.125f;   // 1/sqrt(64)

    // Tile-loader coordinates (256 threads cover both tiles)
    const int kkk = tid >> 3, kc8 = (tid & 7) * 8;   // K tile [32][64]
    const int vdh = tid >> 2, vc4 = (tid & 3) * 8;   // V^T tile [64][32]
    const __bf16* kbase = kw  + (((size_t)b * H_ + h) * LK_ + kkk) * DH_ + kc8;
    const __bf16* vbase = vtw + (((size_t)b * H_ + h) * DH_ + vdh) * LK_ + vc4;

    // Query A-fragments (two K=32 chunks over DH=64), straight from global.
    const __bf16* qrow = qw + (((size_t)b * H_ + h) * LQ_ + qtile + r) * DH_;
    bf16x16 qf0 = load_a_frag(qrow, half);
    bf16x16 qf1 = load_a_frag(qrow + 32, half);

    // Cluster ids for this wave's 8 rows (row = j + 8*half)
    int aq[8];
#pragma unroll
    for (int j = 0; j < 8; ++j)
        aq[j] = assign_q[b * LQ_ + qtile + j + 8 * half];

    float mrun[8], lrun[8];
    f32x8 acc[4];
#pragma unroll
    for (int j = 0; j < 8; ++j) { mrun[j] = -1e30f; lrun[j] = 0.f; }
#pragma unroll
    for (int d = 0; d < 4; ++d) acc[d] = (f32x8){0,0,0,0,0,0,0,0};

    __bf16* pwave = &Pb[wave * 16 * PB_STR];

    for (int kv = 0; kv < LK_; kv += 32) {
        __syncthreads();
        // K tile [32][64] and V^T tile [64][32] via async global->LDS copies
        copy16_g2l(&Kt[kkk * KT_STR + kc8], kbase + (size_t)kv * DH_);
        copy16_g2l(&Vt[vdh * VT_STR + vc4], vbase + kv);
        wait_async_done();
        __syncthreads();

        // Scores: two 16-key groups, each accumulated over two DH chunks.
        f32x8 s[2];
        int   ak[2];
#pragma unroll
        for (int g = 0; g < 2; ++g) {
            f32x8 c = (f32x8){0,0,0,0,0,0,0,0};
            bf16x16 b0 = load_b_frag(&Kt[(g * 16 + r) * KT_STR + 0],  half);
            c = wmma_bf16(qf0, b0, c);
            bf16x16 b1 = load_b_frag(&Kt[(g * 16 + r) * KT_STR + 32], half);
            c = wmma_bf16(qf1, b1, c);
            s[g] = c;
            ak[g] = assign_k[b * LK_ + kv + g * 16 + r];
        }

        // Online softmax with cluster mask (p = 0 outside the cluster).
#pragma unroll
        for (int j = 0; j < 8; ++j) {
            bool v0 = (aq[j] == ak[0]), v1 = (aq[j] == ak[1]);
            float s0 = s[0][j] * SCALE, s1 = s[1][j] * SCALE;
            float t = fmaxf(v0 ? s0 : -1e30f, v1 ? s1 : -1e30f);
            t = rmax16(t);
            float mnew = fmaxf(mrun[j], t);
            float f = __expf(mrun[j] - mnew);
            mrun[j] = mnew;
            float p0 = v0 ? __expf(s0 - mnew) : 0.f;
            float p1 = v1 ? __expf(s1 - mnew) : 0.f;
            lrun[j] = lrun[j] * f + rsum16(p0 + p1);
#pragma unroll
            for (int d = 0; d < 4; ++d) acc[d][j] *= f;
            int prow = j + 8 * half;
            pwave[prow * PB_STR + r]      = f2bf(p0);
            pwave[prow * PB_STR + 16 + r] = f2bf(p1);
        }
        // wave-internal LDS visibility for the C-layout -> A-layout transpose
        asm volatile("s_wait_dscnt 0" ::: "memory");

        bf16x16 pfrag = load_a_frag(&pwave[r * PB_STR], half);
#pragma unroll
        for (int d = 0; d < 4; ++d) {
            bf16x16 bfrag = load_b_frag(&Vt[(d * 16 + r) * VT_STR], half);
            acc[d] = wmma_bf16(pfrag, bfrag, acc[d]);
        }
    }

    // Normalize; rows with no in-cluster key (lrun==0) output exact zero.
#pragma unroll
    for (int j = 0; j < 8; ++j) {
        float inv = (lrun[j] > 0.f) ? 1.f / lrun[j] : 0.f;
        int qg = qtile + j + 8 * half;
#pragma unroll
        for (int d = 0; d < 4; ++d) {
            int col = h * DH_ + d * 16 + r;
            ctx[((size_t)b * LQ_ + qg) * D_ + col] = acc[d][j] * inv;
        }
    }
}

// ---------------------------------------------------------------------------
// Host launcher
// ---------------------------------------------------------------------------
extern "C" void kernel_launch(void* const* d_in, const int* in_sizes, int n_in,
                              void* d_out, int out_size, void* d_ws, size_t ws_size,
                              hipStream_t stream) {
    const float* queries = (const float*)d_in[0];
    const float* keys    = (const float*)d_in[1];
    const float* values  = (const float*)d_in[2];
    const float* Wq      = (const float*)d_in[3];
    const float* Wk      = (const float*)d_in[4];
    const float* Wv      = (const float*)d_in[5];
    const float* Wo      = (const float*)d_in[6];
    const float* Wr      = (const float*)d_in[7];
    float* out = (float*)d_out;

    // Workspace carve-up
    char* ws = (char*)d_ws;
    const size_t headElems = (size_t)B_ * H_ * LQ_ * DH_;     // 4M elems
    __bf16* q_ws  = (__bf16*)ws;                 ws += headElems * 2;  // 8 MiB
    __bf16* k_ws  = (__bf16*)ws;                 ws += headElems * 2;
    __bf16* vT_ws = (__bf16*)ws;                 ws += headElems * 2;
    float*  ctx_ws = (float*)ws;                 ws += (size_t)B_ * LQ_ * D_ * 4;
    int* assign_q = (int*)ws;                    ws += (size_t)B_ * LQ_ * 4;
    int* assign_k = (int*)ws;

    const int tokens = B_ * LQ_;

    // 1) routing
    router_kernel<<<(tokens + 255) / 256, 256, 0, stream>>>(queries, Wr, assign_q, tokens);
    router_kernel<<<(tokens + 255) / 256, 256, 0, stream>>>(keys,    Wr, assign_k, tokens);

    // 2) projections (WMMA bf16)
    dim3 ggrid(D_ / 64, tokens / 128);
    wmma_gemm_kernel<0><<<ggrid, 256, 0, stream>>>(queries, Wq, q_ws);
    wmma_gemm_kernel<0><<<ggrid, 256, 0, stream>>>(keys,    Wk, k_ws);
    wmma_gemm_kernel<1><<<ggrid, 256, 0, stream>>>(values,  Wv, vT_ws);

    // 3) clustered flash attention
    dim3 agrid(LQ_ / 128, H_, B_);
    attention_kernel<<<agrid, 256, 0, stream>>>(q_ws, k_ws, vT_ws, assign_q, assign_k, ctx_ws);

    // 4) output projection
    wmma_gemm_kernel<2><<<ggrid, 256, 0, stream>>>(ctx_ws, Wo, out);
}
